// GNNLayer_33741263077794
// MI455X (gfx1250) — matrix operand order repaired and checked by reading, not compiled, not run on hardware
//
#include <hip/hip_runtime.h>
#include <hip/hip_bf16.h>
#include <math.h>

// ---------------------------------------------------------------------------
// GatedGCN layer, B=4 V=256 H=128, fp32 in/out, f16 WMMA GEMMs (f32 accum)
// Edge weights staged in LDS (padded, bank-conflict-free) for WMMA B-frags.
// ---------------------------------------------------------------------------

typedef __attribute__((ext_vector_type(16))) _Float16 v16h;
typedef __attribute__((ext_vector_type(8)))  float    v8f;

#define BB 4
#define VV 256
#define HH 128
#define EDGE_ROWS (BB*VV*VV)               // 262144
#define EDGE_ELEMS ((size_t)EDGE_ROWS*HH)  // 33554432
#define NODE_ELEMS (BB*VV*HH)              // 131072
#define NBLK (EDGE_ROWS/16)                // 16384
#define WSTRIDE 136                        // padded LDS row stride (halves): 272B,
                                           // 68 dwords -> lane l2 maps to dwords
                                           // 4*l2..4*l2+3 : conflict-free b128 reads
#define WMAT (HH*WSTRIDE)                  // halves per staged matrix

// --- A fragment: 16x32 f16 from fp32 row-major x (ld=128), K-chunk base k0.
// 16-bit A 16x32: lane l (l2=l&15 row, hi=l>>4):
//   halves 0..15 -> K = k0 + hi*16 + h (contiguous, matches B frags)
__device__ inline v16h load_afrag(const float* __restrict__ xblk, int k0, int lane) {
    int l2 = lane & 15, hi = lane >> 4;
    const float* p = xblk + (size_t)l2 * HH + k0 + hi * 16;
    v16h a;
#pragma unroll
    for (int h = 0; h < 8; ++h) {
        a[h]     = (_Float16)p[h];
        a[8 + h] = (_Float16)p[8 + h];
    }
    return a;
}

// --- B fragment from global f16 weights (row-major, ld=128)
__device__ inline v16h load_bfrag_g(const _Float16* __restrict__ W, int n0, int k0, int lane) {
    int l2 = lane & 15, hi = lane >> 4;
    const _Float16* p = W + (size_t)(n0 + l2) * HH + k0 + hi * 16;
    v16h b;
#pragma unroll
    for (int h = 0; h < 16; ++h) b[h] = p[h];
    return b;
}

// --- B fragment from LDS-staged f16 weights (row stride WSTRIDE halves)
__device__ inline v16h load_bfrag_lds(const _Float16* W, int n0, int k0, int lane) {
    int l2 = lane & 15, hi = lane >> 4;
    const _Float16* p = W + (n0 + l2) * WSTRIDE + k0 + hi * 16;
    v16h b;
#pragma unroll
    for (int h = 0; h < 16; ++h) b[h] = p[h];
    return b;
}

// 16x16 output tile of x(16x128) @ W^T, K=128 in 4 chunks of 32 (global B)
__device__ inline v8f wmma_tile_g(const v16h* A, const _Float16* __restrict__ W,
                                  int n0, int lane) {
    v8f c = {};
#pragma unroll
    for (int kc = 0; kc < 4; ++kc) {
        v16h b = load_bfrag_g(W, n0, kc * 32, lane);
        c = __builtin_amdgcn_wmma_f32_16x16x32_f16(false, A[kc], false, b,
                                                   (short)0, c, false, false);
    }
    return c;
}

// same with LDS-resident B
__device__ inline v8f wmma_tile_l(const v16h* A, const _Float16* W, int n0, int lane) {
    v8f c = {};
#pragma unroll
    for (int kc = 0; kc < 4; ++kc) {
        v16h b = load_bfrag_lds(W, n0, kc * 32, lane);
        c = __builtin_amdgcn_wmma_f32_16x16x32_f16(false, A[kc], false, b,
                                                   (short)0, c, false, false);
    }
    return c;
}

// ---------------------------------------------------------------------------
// Kernel 0: convert 8 weight matrices (128x128 f32) -> f16, contiguous
//           order: WU, WV, WA, WB, WC, WD, WUe, WVe
// ---------------------------------------------------------------------------
__global__ void cvt_weights_kernel(const float* w0, const float* w1, const float* w2,
                                   const float* w3, const float* w4, const float* w5,
                                   const float* w6, const float* w7,
                                   _Float16* __restrict__ out) {
    int i = blockIdx.x * blockDim.x + threadIdx.x;
    if (i >= 8 * HH * HH) return;
    int m = i >> 14, r = i & 16383;
    const float* w;
    switch (m) {
        case 0: w = w0; break; case 1: w = w1; break;
        case 2: w = w2; break; case 3: w = w3; break;
        case 4: w = w4; break; case 5: w = w5; break;
        case 6: w = w6; break; default: w = w7; break;
    }
    out[i] = (_Float16)w[r];
}

// ---------------------------------------------------------------------------
// Kernel 1: node GEMMs  Uh/Vh/Ah/Bh = h @ W^T + b   (1024x128 each)
// ---------------------------------------------------------------------------
__global__ void gemm_h_kernel(const float* __restrict__ x, const _Float16* __restrict__ Wall,
                              const float* bU, const float* bV, const float* bA, const float* bB,
                              float* Uh, float* Vh, float* Ah, float* Bh) {
    int wid  = (blockIdx.x * blockDim.x + threadIdx.x) >> 5;
    int lane = threadIdx.x & 31;
    if (wid >= 64 * 4) return;
    int mat = wid & 3, rb = wid >> 2;
    const float* bias; float* out;
    switch (mat) {
        case 0: bias = bU; out = Uh; break;
        case 1: bias = bV; out = Vh; break;
        case 2: bias = bA; out = Ah; break;
        default: bias = bB; out = Bh; break;
    }
    const _Float16* W = Wall + (size_t)mat * HH * HH;
    const float* xb = x + (size_t)rb * 16 * HH;
    v16h A[4];
#pragma unroll
    for (int kc = 0; kc < 4; ++kc) A[kc] = load_afrag(xb, kc * 32, lane);
    int l2 = lane & 15, hi = lane >> 4;
#pragma unroll
    for (int t = 0; t < 8; ++t) {
        int n0 = t * 16;
        v8f c = wmma_tile_g(A, W, n0, lane);
        float bv = bias[n0 + l2];
#pragma unroll
        for (int r = 0; r < 8; ++r)
            out[(size_t)(rb * 16 + r + hi * 8) * HH + n0 + l2] = c[r] + bv;
    }
}

// ---------------------------------------------------------------------------
// Kernel 2: fused edge pipeline (one wave per 16-edge-row block, grid-stride).
//   Ce = e@WC^T, Dew = ew@WD^T, Vew = ew@WVe^T, Uew = ew@WUe^T  (WMMA, B in LDS)
//   ew_pre = (Ah[j]+Bh[i]+Dew+bD)*g ; e_new = ew_pre + (Ce+bC)*g
//   gate = sigmoid(e_new) ; agg[b,i,:] += gate*(Vh[j]+Vew+bVe)*g
//   ew_new = ew_pre + Uew + bUe
// ---------------------------------------------------------------------------
__global__ void edge_kernel(const float* __restrict__ e, const float* __restrict__ ew,
                            const int* __restrict__ graph, const _Float16* __restrict__ Wall,
                            const float* __restrict__ bC, const float* __restrict__ bD,
                            const float* __restrict__ bUe, const float* __restrict__ bVe,
                            const float* __restrict__ Ah, const float* __restrict__ Bh,
                            const float* __restrict__ Vh,
                            float* __restrict__ eN, float* __restrict__ ewN,
                            float* __restrict__ agg, float* __restrict__ sums) {
    __shared__ _Float16 sW[4 * WMAT];     // WC, WD, WUe, WVe (padded rows)
    __shared__ float s_acc[4 * HH];
    for (int i = threadIdx.x; i < 4 * HH; i += blockDim.x) s_acc[i] = 0.0f;
    // stage 4 edge weight mats (Wall mats 4..7) into padded LDS, 16B chunks
    for (int idx = threadIdx.x; idx < 4 * HH * 16; idx += blockDim.x) {
        int row = idx >> 4, seg = idx & 15;     // row in 0..511, seg in 0..15
        int m = row >> 7, r = row & 127;
        const uint4* src = (const uint4*)(Wall + (size_t)(4 + m) * HH * HH + r * HH);
        *(uint4*)(sW + m * WMAT + r * WSTRIDE + seg * 8) = src[seg];
    }
    __syncthreads();

    const _Float16* sWC  = sW;
    const _Float16* sWD  = sW + WMAT;
    const _Float16* sWUe = sW + 2 * WMAT;
    const _Float16* sWVe = sW + 3 * WMAT;

    int lane = threadIdx.x & 31, l2 = lane & 15, hi = lane >> 4;
    int wid    = (blockIdx.x * blockDim.x + threadIdx.x) >> 5;
    int nwaves = (gridDim.x * blockDim.x) >> 5;

    for (int blk = wid; blk < NBLK; blk += nwaves) {
        // prefetch next grid-stride block's activation rows (8KB each tensor)
        int nxt = blk + nwaves;
        if (nxt < NBLK) {
            __builtin_prefetch(e  + (size_t)nxt * 16 * HH + lane * 64, 0, 1);
            __builtin_prefetch(ew + (size_t)nxt * 16 * HH + lane * 64, 0, 1);
        }

        int start = blk * 16;                   // flattened edge row
        int b   = start >> 16;                  // / (V*V)
        int rem = start & 65535;
        int ni  = rem >> 8;                     // node i
        int j0  = rem & 255;                    // node j block start (mult of 16)

        v16h Ae[4], Aw[4];
#pragma unroll
        for (int kc = 0; kc < 4; ++kc) {
            Ae[kc] = load_afrag(e  + (size_t)start * HH, kc * 32, lane);
            Aw[kc] = load_afrag(ew + (size_t)start * HH, kc * 32, lane);
        }
        float gv[8];
        int gbase = (b * VV + ni) * VV + j0 + hi * 8;
#pragma unroll
        for (int r = 0; r < 8; ++r) gv[r] = (float)graph[gbase + r];

        int hrow = (b * VV + ni) * HH;

        for (int t = 0; t < 8; ++t) {
            int n0 = t * 16;
            v8f Cc = wmma_tile_l(Ae, sWC,  n0, lane);
            v8f Dc = wmma_tile_l(Aw, sWD,  n0, lane);
            v8f Vc = wmma_tile_l(Aw, sWVe, n0, lane);
            v8f Uc = wmma_tile_l(Aw, sWUe, n0, lane);

            int nl = n0 + l2;
            float bCv = bC[nl], bDv = bD[nl], bVev = bVe[nl], bUev = bUe[nl];
            float Bhv = Bh[hrow + nl];

            float aggp = 0.f, se = 0.f, se2 = 0.f, sw = 0.f, sw2 = 0.f;
#pragma unroll
            for (int r = 0; r < 8; ++r) {
                int m  = r + hi * 8;
                int jr = j0 + m;
                float ahv = Ah[(size_t)(b * VV + jr) * HH + nl];
                float vhv = Vh[(size_t)(b * VV + jr) * HH + nl];
                float g = gv[r];
                float ewpre = (ahv + Bhv + Dc[r] + bDv) * g;
                float enew  = ewpre + (Cc[r] + bCv) * g;
                float gate  = 1.0f / (1.0f + __expf(-enew));
                aggp += gate * (vhv + Vc[r] + bVev) * g;
                float ewnew = ewpre + Uc[r] + bUev;
                size_t orow = (size_t)(start + m) * HH + nl;
                eN[orow]  = enew;
                ewN[orow] = ewnew;
                se += enew; se2 += enew * enew;
                sw += ewnew; sw2 += ewnew * ewnew;
            }
            // combine both half-wave row groups, one atomic per channel
            aggp += __shfl_xor(aggp, 16, 32);
            if (hi == 0) atomicAdd(&agg[hrow + nl], aggp);
            atomicAdd(&s_acc[nl],          se);
            atomicAdd(&s_acc[HH + nl],     se2);
            atomicAdd(&s_acc[2 * HH + nl], sw);
            atomicAdd(&s_acc[3 * HH + nl], sw2);
        }
    }
    __syncthreads();
    for (int i = threadIdx.x; i < HH; i += blockDim.x) {
        atomicAdd(&sums[2 * HH + i], s_acc[i]);            // sum e
        atomicAdd(&sums[3 * HH + i], s_acc[HH + i]);       // sumsq e
        atomicAdd(&sums[4 * HH + i], s_acc[2 * HH + i]);   // sum ew
        atomicAdd(&sums[5 * HH + i], s_acc[3 * HH + i]);   // sumsq ew
    }
}

// ---------------------------------------------------------------------------
// Kernel 3: h_new = Uh + agg ; accumulate h channel stats
// ---------------------------------------------------------------------------
__global__ void hnew_kernel(const float* __restrict__ Uh, const float* __restrict__ agg,
                            float* __restrict__ hN, float* __restrict__ sums) {
    int c = threadIdx.x;   // 128 threads
    float s = 0.f, s2 = 0.f;
    for (int r = blockIdx.x; r < BB * VV; r += gridDim.x) {
        float v = Uh[(size_t)r * HH + c] + agg[(size_t)r * HH + c];
        hN[(size_t)r * HH + c] = v;
        s += v; s2 += v * v;
    }
    atomicAdd(&sums[c], s);
    atomicAdd(&sums[HH + c], s2);
}

// ---------------------------------------------------------------------------
// Kernel 4: BN finalize -> per-channel scale/shift for h, e, ew
// ---------------------------------------------------------------------------
__global__ void finalize_kernel(const float* __restrict__ sums,
                                const float* gh, const float* bh,
                                const float* ge, const float* be,
                                const float* gew, const float* bew,
                                float* __restrict__ aff) {
    int c = threadIdx.x;   // 128
    const float EPS = 1e-5f;
    {   float n = (float)(BB * VV);
        float m = sums[c] / n, var = sums[HH + c] / n - m * m;
        float sc = gh[c] * rsqrtf(var + EPS);
        aff[c] = sc; aff[HH + c] = bh[c] - m * sc; }
    {   float n = (float)EDGE_ROWS;
        float m = sums[2 * HH + c] / n, var = sums[3 * HH + c] / n - m * m;
        float sc = ge[c] * rsqrtf(var + EPS);
        aff[2 * HH + c] = sc; aff[3 * HH + c] = be[c] - m * sc; }
    {   float n = (float)EDGE_ROWS;
        float m = sums[4 * HH + c] / n, var = sums[5 * HH + c] / n - m * m;
        float sc = gew[c] * rsqrtf(var + EPS);
        aff[4 * HH + c] = sc; aff[5 * HH + c] = bew[c] - m * sc; }
}

// ---------------------------------------------------------------------------
// Kernel 5: out = x + relu(xN * scale[c] + shift[c]), float4 vectorized
// ---------------------------------------------------------------------------
__global__ void bnout_kernel(const float* __restrict__ x, const float* __restrict__ xN,
                             const float* __restrict__ scale, const float* __restrict__ shift,
                             float* __restrict__ out, long n4) {
    long i      = (long)blockIdx.x * blockDim.x + threadIdx.x;
    long stride = (long)gridDim.x * blockDim.x;
    const float4* xv  = (const float4*)x;
    const float4* xNv = (const float4*)xN;
    float4*       ov  = (float4*)out;
    for (; i < n4; i += stride) {
        int c = (int)((i * 4) & (HH - 1));
        float4 a = xv[i], bn = xNv[i], o;
        o.x = a.x + fmaxf(0.f, bn.x * scale[c]     + shift[c]);
        o.y = a.y + fmaxf(0.f, bn.y * scale[c + 1] + shift[c + 1]);
        o.z = a.z + fmaxf(0.f, bn.z * scale[c + 2] + shift[c + 2]);
        o.w = a.w + fmaxf(0.f, bn.w * scale[c + 3] + shift[c + 3]);
        ov[i] = o;
    }
}

// ---------------------------------------------------------------------------
extern "C" void kernel_launch(void* const* d_in, const int* in_sizes, int n_in,
                              void* d_out, int out_size, void* d_ws, size_t ws_size,
                              hipStream_t stream) {
    const float* h   = (const float*)d_in[0];
    const float* e   = (const float*)d_in[1];
    const float* ew  = (const float*)d_in[2];
    const float* WU  = (const float*)d_in[3];  const float* bU  = (const float*)d_in[4];
    const float* WV  = (const float*)d_in[5];  const float* bV  = (const float*)d_in[6];
    const float* WA  = (const float*)d_in[7];  const float* bA  = (const float*)d_in[8];
    const float* WB  = (const float*)d_in[9];  const float* bB  = (const float*)d_in[10];
    const float* WC  = (const float*)d_in[11]; const float* bC  = (const float*)d_in[12];
    const float* WD  = (const float*)d_in[13]; const float* bD  = (const float*)d_in[14];
    const float* WUe = (const float*)d_in[15]; const float* bUe = (const float*)d_in[16];
    const float* WVe = (const float*)d_in[17]; const float* bVe = (const float*)d_in[18];
    const float* gamma_h  = (const float*)d_in[19]; const float* beta_h  = (const float*)d_in[20];
    const float* gamma_e  = (const float*)d_in[21]; const float* beta_e  = (const float*)d_in[22];
    const float* gamma_ew = (const float*)d_in[23]; const float* beta_ew = (const float*)d_in[24];
    const int*   graph    = (const int*)d_in[25];

    // workspace layout (floats)
    float* ws   = (float*)d_ws;
    float* eN   = ws;                        // 33554432
    float* ewN  = eN  + EDGE_ELEMS;          // 33554432
    float* Uh   = ewN + EDGE_ELEMS;          // 131072
    float* Vhp  = Uh  + NODE_ELEMS;
    float* Ahp  = Vhp + NODE_ELEMS;
    float* Bhp  = Ahp + NODE_ELEMS;
    float* hN   = Bhp + NODE_ELEMS;
    float* agg  = hN  + NODE_ELEMS;          // zeroed each call
    float* sums = agg + NODE_ELEMS;          // 6*128, zeroed each call
    float* aff  = sums + 6 * HH;             // 6*128
    _Float16* Wf16 = (_Float16*)(aff + 6 * HH);  // 8*128*128 halves

    hipMemsetAsync(agg, 0, (NODE_ELEMS + 6 * HH) * sizeof(float), stream);

    cvt_weights_kernel<<<(8 * HH * HH + 255) / 256, 256, 0, stream>>>(
        WU, WV, WA, WB, WC, WD, WUe, WVe, Wf16);

    gemm_h_kernel<<<32, 256, 0, stream>>>(h, Wf16, bU, bV, bA, bB, Uh, Vhp, Ahp, Bhp);

    edge_kernel<<<512, 256, 0, stream>>>(e, ew, graph, Wf16, bC, bD, bUe, bVe,
                                         Ahp, Bhp, Vhp, eN, ewN, agg, sums);

    hnew_kernel<<<64, HH, 0, stream>>>(Uh, agg, hN, sums);

    finalize_kernel<<<1, HH, 0, stream>>>(sums, gamma_h, beta_h, gamma_e, beta_e,
                                          gamma_ew, beta_ew, aff);

    float* out = (float*)d_out;
    bnout_kernel<<<128, 256, 0, stream>>>(h, hN, aff, aff + HH, out,
                                          (long)NODE_ELEMS / 4);
    bnout_kernel<<<4096, 256, 0, stream>>>(e, eN, aff + 2 * HH, aff + 3 * HH,
                                           out + NODE_ELEMS, (long)EDGE_ELEMS / 4);
    bnout_kernel<<<4096, 256, 0, stream>>>(ew, ewN, aff + 4 * HH, aff + 5 * HH,
                                           out + NODE_ELEMS + EDGE_ELEMS,
                                           (long)EDGE_ELEMS / 4);
}
